// RNNModel_30459908063560
// MI455X (gfx1250) — compile-verified, matrix operand order checked by reading, not run on hardware
//
#include <hip/hip_runtime.h>

// ---- problem constants (from reference) ----
#define LSEQ 256
#define BSZ  32
#define HID  1024
#define NIN  1024
#define VOC  10000
#define GAT  4096   // 4*HID

typedef __attribute__((ext_vector_type(16))) __bf16 v16bf;
typedef __attribute__((ext_vector_type(8)))  __bf16 v8bf;
typedef __attribute__((ext_vector_type(8)))  float  v8f;
typedef __attribute__((ext_vector_type(2)))  float  v2f;

__device__ __forceinline__ float sigf(float x) { return 1.f / (1.f + __expf(-x)); }

// ---- inverse of the CDNA5 wave32 16-bit WMMA fragment layout (ISA 7.12.2) ----
// For a 16x32 (A) or 32x16 (B) tile, element (row/col rc, k) lives at:
//   lane = (rc & 15) | ((k & 8) << 1)
//   e    = ((k >> 4) << 3) | (k & 7)         (e = index into the lane's v16bf)
// Staging tiles in [lane][e] order makes every fragment load a contiguous
// 32-byte LDS vector load (2 x ds_load_b128) instead of 16 scalar ds_load_u16.
__device__ __forceinline__ int frag_lane(int rc, int kl) { return (rc & 15) | ((kl & 8) << 1); }
__device__ __forceinline__ int frag_e0(int kl) { return ((kl >> 4) << 3) | (kl & 7); }

// ============================================================
// f32 -> bf16 conversion, 8 elements/thread
// ============================================================
__global__ __launch_bounds__(256)
void k_f32_to_bf16v(const float* __restrict__ src, __bf16* __restrict__ dst, int n8) {
  const int i = blockIdx.x * 256 + threadIdx.x;
  if (i < n8) {
    const float4 a = ((const float4*)src)[2 * i];
    const float4 b = ((const float4*)src)[2 * i + 1];
    v8bf o;
    o[0] = (__bf16)a.x; o[1] = (__bf16)a.y; o[2] = (__bf16)a.z; o[3] = (__bf16)a.w;
    o[4] = (__bf16)b.x; o[5] = (__bf16)b.y; o[6] = (__bf16)b.z; o[7] = (__bf16)b.w;
    ((v8bf*)dst)[i] = o;
  }
}

// ============================================================
// gates_x = gather(emb_w, ids) @ w_ih^T   (bf16 WMMA, f32 acc)
// block tile 32(M) x 64(N), KC=64, double-buffered LDS in fragment order
// grid: (GAT/64, L*B/32), block 256 (8 waves -> 2x4 of 16x16 tiles)
// ============================================================
__global__ __launch_bounds__(256)
void k_gemm_gates_x(const int* __restrict__ ids, const __bf16* __restrict__ emb,
                    const __bf16* __restrict__ wih, float* __restrict__ gx) {
  __shared__ __align__(32) __bf16 Af[2][2][2][32][16];  // [buf][mtile][kslice][lane][e]
  __shared__ __align__(32) __bf16 Bf[2][4][2][32][16];  // [buf][ntile][kslice][lane][e]
  __shared__ int rowid[32];
  const int tid = threadIdx.x;
  const int n0 = blockIdx.x * 64, m0 = blockIdx.y * 32;
  if (tid < 32) rowid[tid] = ids[m0 + tid];
  __syncthreads();
  const int wave = tid >> 5, lane = tid & 31;
  const int mt = wave & 1, nt = wave >> 1;
  const int ar = tid >> 3, ac = (tid & 7) * 8;   // A staging: row, k-offset (8 bf16)
  const int bn = tid >> 2, bk = (tid & 3) * 16;  // B staging: col, k-offset (16 bf16)
  const size_t arow = (size_t)rowid[ar] * NIN;
  const size_t brow = (size_t)(n0 + bn) * NIN;

  auto stage = [&](int buf, int kk) {
    {
      const int ks = ac >> 5, kl = ac & 31;
      *(v8bf*)&Af[buf][ar >> 4][ks][frag_lane(ar, kl)][frag_e0(kl)] =
          *(const v8bf*)(emb + arow + kk + ac);
    }
#pragma unroll
    for (int h = 0; h < 2; ++h) {
      const int k = bk + 8 * h, ks = k >> 5, kl = k & 31;
      *(v8bf*)&Bf[buf][bn >> 4][ks][frag_lane(bn, kl)][frag_e0(kl)] =
          *(const v8bf*)(wih + brow + kk + k);
    }
  };

  v8f acc = {};
  stage(0, 0);
  __syncthreads();
  const int NIT = NIN / 64;
  for (int it = 0; it < NIT; ++it) {
    const int cur = it & 1;
    if (it + 1 < NIT) stage(1 - cur, (it + 1) * 64);
    if (it + 2 < NIT) {
      __builtin_prefetch(emb + arow + (it + 2) * 64 + ac, 0, 3);
      __builtin_prefetch(wih + brow + (it + 2) * 64 + bk, 0, 3);
    }
#pragma unroll
    for (int ks = 0; ks < 2; ++ks) {
      v16bf a = *(const v16bf*)&Af[cur][mt][ks][lane][0];
      v16bf b = *(const v16bf*)&Bf[cur][nt][ks][lane][0];
      acc = __builtin_amdgcn_wmma_f32_16x16x32_bf16(false, a, false, b, (short)0, acc,
                                                    false, false);
    }
    __syncthreads();
  }
  const int lm = (lane & 16) ? 8 : 0, ln = lane & 15;
#pragma unroll
  for (int r = 0; r < 8; ++r)
    gx[(size_t)(m0 + mt * 16 + lm + r) * GAT + (n0 + nt * 16 + ln)] = acc[r];
}

// ============================================================
// gates = gates_x[t] + h @ w_hh^T   (f32 WMMA 16x16x4, full precision recurrence)
// KC=32, double-buffered; B staged pair-interleaved so each K-pair fragment
// is one ds_load_b64. grid: GAT/64, block 256
// ============================================================
__global__ __launch_bounds__(256)
void k_lstm_gemm(const float* __restrict__ gx_t, const float* __restrict__ h,
                 const float* __restrict__ whh, float* __restrict__ gates) {
  __shared__ float As[2][32][36];      // row-major, pairs contiguous
  __shared__ float Bp[2][16][64][2];   // [buf][k/2][n][k&1]
  const int tid = threadIdx.x;
  const int n0 = blockIdx.x * 64;
  const int wave = tid >> 5, lane = tid & 31;
  const int mw = (wave & 1) * 16, nw = (wave >> 1) * 16;
  const int lm = (lane & 16) ? 8 : 0, ln = lane & 15;
  const int ka = (lane & 16) ? 2 : 0;
  const int ar = tid >> 3, ac = (tid & 7) * 4;
  const int bn = tid >> 2, bk = (tid & 3) * 8;
  const size_t brow = (size_t)(n0 + bn) * HID;

  auto stage = [&](int buf, int kk) {
    *(float4*)&As[buf][ar][ac] = *(const float4*)(h + (size_t)ar * HID + kk + ac);
#pragma unroll
    for (int i = 0; i < 8; i += 2) {
      const int k = bk + i;
      *(float2*)&Bp[buf][k >> 1][bn][0] = *(const float2*)(whh + brow + kk + k);
    }
  };

  v8f acc;
#pragma unroll
  for (int r = 0; r < 8; ++r)
    acc[r] = gx_t[(size_t)(mw + lm + r) * GAT + (n0 + nw + ln)];

  stage(0, 0);
  __syncthreads();
  const int NIT = HID / 32;
  for (int it = 0; it < NIT; ++it) {
    const int cur = it & 1;
    if (it + 1 < NIT) stage(1 - cur, (it + 1) * 32);
    if (it + 2 < NIT) __builtin_prefetch(whh + brow + (it + 2) * 32 + bk, 0, 3);
#pragma unroll
    for (int kc = 0; kc < 32; kc += 4) {
      v2f a = *(const v2f*)&As[cur][mw + ln][kc + ka];
      v2f b = *(const v2f*)&Bp[cur][(kc + ka) >> 1][nw + ln][0];
      acc = __builtin_amdgcn_wmma_f32_16x16x4_f32(false, a, false, b, (short)0, acc,
                                                  false, false);
    }
    __syncthreads();
  }
#pragma unroll
  for (int r = 0; r < 8; ++r)
    gates[(size_t)(mw + lm + r) * GAT + (n0 + nw + ln)] = acc[r];
}

// ============================================================
// LSTM pointwise: c,h update; writes h into hist[t]
// ============================================================
__global__ __launch_bounds__(256)
void k_lstm_point(const float* __restrict__ gates, float* __restrict__ hbuf,
                  float* __restrict__ cbuf, float* __restrict__ hist_t) {
  const int idx = blockIdx.x * 256 + threadIdx.x;  // < BSZ*HID
  const int b = idx >> 10;
  const int j = idx & (HID - 1);
  const float gi = gates[(size_t)b * GAT + j];
  const float gf = gates[(size_t)b * GAT + HID + j];
  const float gg = gates[(size_t)b * GAT + 2 * HID + j];
  const float go = gates[(size_t)b * GAT + 3 * HID + j];
  const float c = sigf(gf) * cbuf[idx] + sigf(gi) * tanhf(gg);
  const float hh = sigf(go) * tanhf(c);
  cbuf[idx] = c;
  hbuf[idx] = hh;
  hist_t[idx] = hh;
}

// ============================================================
// Attention for step t: one block per batch element, one score per thread
// (L==256==blockDim). Caches bU[t]=h_t.U incrementally.
// ============================================================
__global__ __launch_bounds__(256)
void k_attention(const float* __restrict__ hist, const float* __restrict__ Wv,
                 const float* __restrict__ Uv, float* __restrict__ bU,
                 float* __restrict__ ctx, int t) {
  __shared__ float redA[256], redB[256], att[256];
  const int b = blockIdx.x, tid = threadIdx.x;
  const float* ht = hist + ((size_t)t * BSZ + b) * HID;
  float sa = 0.f, su = 0.f;
  for (int j = tid; j < HID; j += 256) {
    const float hv = ht[j];
    sa += hv * Wv[j];
    su += hv * Uv[j];
  }
  redA[tid] = sa; redB[tid] = su;
  __syncthreads();
  for (int s = 128; s > 0; s >>= 1) {
    if (tid < s) { redA[tid] += redA[tid + s]; redB[tid] += redB[tid + s]; }
    __syncthreads();
  }
  if (tid == 0) bU[(size_t)t * BSZ + b] = redB[0];
  const float a = redA[0];
  __syncthreads();
  float sc = -3.0e38f;
  if (tid <= t) sc = tanhf(a + bU[(size_t)tid * BSZ + b]);
  redA[tid] = sc;
  __syncthreads();
  for (int s = 128; s > 0; s >>= 1) {
    if (tid < s) redA[tid] = fmaxf(redA[tid], redA[tid + s]);
    __syncthreads();
  }
  const float mx = redA[0];
  __syncthreads();
  const float e = (tid <= t) ? __expf(sc - mx) : 0.f;
  redB[tid] = e;
  __syncthreads();
  for (int s = 128; s > 0; s >>= 1) {
    if (tid < s) redB[tid] += redB[tid + s];
    __syncthreads();
  }
  att[tid] = e / redB[0];
  __syncthreads();
  float4 acc4 = {0.f, 0.f, 0.f, 0.f};
  for (int s = 0; s <= t; ++s) {
    const float w = att[s];
    const float4 hv = ((const float4*)(hist + ((size_t)s * BSZ + b) * HID))[tid];
    acc4.x += w * hv.x; acc4.y += w * hv.y; acc4.z += w * hv.z; acc4.w += w * hv.w;
  }
  ((float4*)(ctx + ((size_t)t * BSZ + b) * HID))[tid] = acc4;
}

// ============================================================
// logits = ctx @ dec_w^T + dec_b   (bf16 WMMA), written straight into d_out
// same double-buffered fragment-order staging; A converted f32->bf16 on the fly
// grid: ((VOC+63)/64, L*B/32)
// ============================================================
__global__ __launch_bounds__(256)
void k_gemm_decoder(const float* __restrict__ ctx, const __bf16* __restrict__ dec,
                    const float* __restrict__ bias, float* __restrict__ out) {
  __shared__ __align__(32) __bf16 Af[2][2][2][32][16];
  __shared__ __align__(32) __bf16 Bf[2][4][2][32][16];
  const int tid = threadIdx.x;
  const int n0 = blockIdx.x * 64, m0 = blockIdx.y * 32;
  const int wave = tid >> 5, lane = tid & 31;
  const int mt = wave & 1, nt = wave >> 1;
  const int ar = tid >> 3, ac = (tid & 7) * 8;
  const int bn = tid >> 2, bk = (tid & 3) * 16;
  const int v = n0 + bn;
  const int vc = (v < VOC) ? v : (VOC - 1);
  const size_t arow = (size_t)(m0 + ar) * HID;
  const size_t brow = (size_t)vc * HID;

  auto stage = [&](int buf, int kk) {
    {
      const int ks = ac >> 5, kl = ac & 31;
      const float* s = ctx + arow + kk + ac;
      v8bf o;
#pragma unroll
      for (int i = 0; i < 8; ++i) o[i] = (__bf16)s[i];
      *(v8bf*)&Af[buf][ar >> 4][ks][frag_lane(ar, kl)][frag_e0(kl)] = o;
    }
#pragma unroll
    for (int h = 0; h < 2; ++h) {
      const int k = bk + 8 * h, ks = k >> 5, kl = k & 31;
      v8bf o;
      if (v < VOC) o = *(const v8bf*)(dec + brow + kk + k);
      else { v8bf z = {}; o = z; }
      *(v8bf*)&Bf[buf][bn >> 4][ks][frag_lane(bn, kl)][frag_e0(kl)] = o;
    }
  };

  v8f acc = {};
  stage(0, 0);
  __syncthreads();
  const int NIT = HID / 64;
  for (int it = 0; it < NIT; ++it) {
    const int cur = it & 1;
    if (it + 1 < NIT) stage(1 - cur, (it + 1) * 64);
    if (it + 2 < NIT) {
      __builtin_prefetch(ctx + arow + (it + 2) * 64 + ac, 0, 3);
      __builtin_prefetch(dec + brow + (it + 2) * 64 + bk, 0, 3);
    }
#pragma unroll
    for (int ks = 0; ks < 2; ++ks) {
      v16bf a = *(const v16bf*)&Af[cur][mt][ks][lane][0];
      v16bf b = *(const v16bf*)&Bf[cur][nt][ks][lane][0];
      acc = __builtin_amdgcn_wmma_f32_16x16x32_bf16(false, a, false, b, (short)0, acc,
                                                    false, false);
    }
    __syncthreads();
  }
  const int lm = (lane & 16) ? 8 : 0, ln = lane & 15;
#pragma unroll
  for (int r = 0; r < 8; ++r) {
    const int m = m0 + mt * 16 + lm + r;
    const int n = n0 + nt * 16 + ln;
    if (n < VOC) out[(size_t)m * VOC + n] = acc[r] + bias[n];
  }
}

// ============================================================
// In-place log_softmax over VOC per row of d_out (VOC = 2500 float4s exactly)
// ============================================================
__global__ __launch_bounds__(256)
void k_log_softmax(float* __restrict__ out) {
  __shared__ float red[256];
  float4* x = (float4*)(out + (size_t)blockIdx.x * VOC);
  const int N4 = VOC / 4;
  const int tid = threadIdx.x;
  float mx = -3.0e38f;
  for (int i = tid; i < N4; i += 256) {
    const float4 a = x[i];
    mx = fmaxf(mx, fmaxf(fmaxf(a.x, a.y), fmaxf(a.z, a.w)));
  }
  red[tid] = mx;
  __syncthreads();
  for (int s = 128; s > 0; s >>= 1) {
    if (tid < s) red[tid] = fmaxf(red[tid], red[tid + s]);
    __syncthreads();
  }
  mx = red[0];
  __syncthreads();
  float sum = 0.f;
  for (int i = tid; i < N4; i += 256) {
    const float4 a = x[i];
    sum += __expf(a.x - mx) + __expf(a.y - mx) + __expf(a.z - mx) + __expf(a.w - mx);
  }
  red[tid] = sum;
  __syncthreads();
  for (int s = 128; s > 0; s >>= 1) {
    if (tid < s) red[tid] += red[tid + s];
    __syncthreads();
  }
  const float ls = mx + logf(red[0]);
  for (int i = tid; i < N4; i += 256) {
    float4 a = x[i];
    a.x -= ls; a.y -= ls; a.z -= ls; a.w -= ls;
    x[i] = a;
  }
}

// ============================================================
extern "C" void kernel_launch(void* const* d_in, const int* in_sizes, int n_in,
                              void* d_out, int out_size, void* d_ws, size_t ws_size,
                              hipStream_t stream) {
  (void)in_sizes; (void)n_in; (void)out_size; (void)ws_size;
  const int*   ids   = (const int*)d_in[0];
  const float* h0    = (const float*)d_in[1];
  const float* c0    = (const float*)d_in[2];
  const float* emb_w = (const float*)d_in[3];
  const float* w_ih  = (const float*)d_in[4];
  const float* w_hh  = (const float*)d_in[5];
  const float* dec_w = (const float*)d_in[6];
  const float* dec_b = (const float*)d_in[7];
  const float* Wv    = (const float*)d_in[8];
  const float* Uv    = (const float*)d_in[9];

  float* out  = (float*)d_out;                     // [L*B, VOC] log-probs
  float* hbuf = out + (size_t)LSEQ * BSZ * VOC;    // live h, ends as output h
  float* cbuf = hbuf + (size_t)BSZ * HID;          // live c, ends as output c

  // workspace layout
  float* gx    = (float*)d_ws;                          // [L*B, GAT]
  float* hist  = gx + (size_t)LSEQ * BSZ * GAT;         // [L, B, HID]
  float* ctx   = hist + (size_t)LSEQ * BSZ * HID;       // [L, B, HID]
  float* gates = ctx + (size_t)LSEQ * BSZ * HID;        // [B, GAT]
  float* bU    = gates + (size_t)BSZ * GAT;             // [L, B]
  __bf16* emb_bf = (__bf16*)(bU + (size_t)LSEQ * BSZ);  // [VOC, NIN]
  __bf16* wih_bf = emb_bf + (size_t)VOC * NIN;          // [GAT, NIN]
  __bf16* dec_bf = wih_bf + (size_t)GAT * NIN;          // [VOC, HID]

  hipMemcpyAsync(hbuf, h0, sizeof(float) * BSZ * HID, hipMemcpyDeviceToDevice, stream);
  hipMemcpyAsync(cbuf, c0, sizeof(float) * BSZ * HID, hipMemcpyDeviceToDevice, stream);

  k_f32_to_bf16v<<<(VOC * NIN / 8 + 255) / 256, 256, 0, stream>>>(emb_w, emb_bf,
                                                                  VOC * NIN / 8);
  k_f32_to_bf16v<<<(GAT * NIN / 8 + 255) / 256, 256, 0, stream>>>(w_ih, wih_bf,
                                                                  GAT * NIN / 8);
  k_f32_to_bf16v<<<(VOC * HID / 8 + 255) / 256, 256, 0, stream>>>(dec_w, dec_bf,
                                                                  VOC * HID / 8);

  // parallel input projection for all timesteps (with fused embedding gather)
  k_gemm_gates_x<<<dim3(GAT / 64, (LSEQ * BSZ) / 32), 256, 0, stream>>>(
      ids, emb_bf, wih_bf, gx);

  // sequential recurrence + attention
  for (int t = 0; t < LSEQ; ++t) {
    k_lstm_gemm<<<dim3(GAT / 64), 256, 0, stream>>>(
        gx + (size_t)t * BSZ * GAT, hbuf, w_hh, gates);
    k_lstm_point<<<(BSZ * HID) / 256, 256, 0, stream>>>(
        gates, hbuf, cbuf, hist + (size_t)t * BSZ * HID);
    k_attention<<<BSZ, 256, 0, stream>>>(hist, Wv, Uv, bU, ctx, t);
  }

  // decoder + in-place log-softmax
  k_gemm_decoder<<<dim3((VOC + 63) / 64, (LSEQ * BSZ) / 32), 256, 0, stream>>>(
      ctx, dec_bf, dec_b, out);
  k_log_softmax<<<LSEQ * BSZ, 256, 0, stream>>>(out);
}
